// MooreS4Block_34127810134322
// MI455X (gfx1250) — matrix operand order repaired and checked by reading, not compiled, not run on hardware
//
#include <hip/hip_runtime.h>
#include <math.h>

// Shapes from the reference
#define B_ 8
#define L_ 2048
#define H_ 512
#define N_ 64
#define F_ 2048
#define T_ 64                 // S4D chunk length
#define C_ (L_ / T_)          // 32 chunks per sequence

typedef __attribute__((ext_vector_type(16))) __bf16 v16bf;
typedef __attribute__((ext_vector_type(8)))  float  v8f;

union Frag16 { v16bf v; uint4 u[2]; };

__device__ __forceinline__ unsigned short f2bf(float x) {
    union { float f; unsigned u; } c; c.f = x;
    unsigned r = c.u + 0x7FFFu + ((c.u >> 16) & 1u);
    return (unsigned short)(r >> 16);
}

__device__ __forceinline__ float gelu_exact(float v) {
    return 0.5f * v * (1.0f + erff(v * 0.70710678118654752f));
}

// CDNA5 async global->LDS copy (16B per lane), tracked by ASYNCcnt.
// The low 32 bits of a flat pointer into LDS are the LDS byte offset.
__device__ __forceinline__ void async_copy_b128(void* lds_dst, const void* gsrc) {
    unsigned int lds = (unsigned int)(unsigned long long)(uintptr_t)lds_dst;
    unsigned long long g = (unsigned long long)(uintptr_t)gsrc;
    asm volatile("global_load_async_to_lds_b128 %0, %1, off"
                 :: "v"(lds), "v"(g) : "memory");
}
__device__ __forceinline__ void async_wait0() {
    asm volatile("s_wait_asynccnt 0x0" ::: "memory");
}

// ---------------------------------------------------------------------------
// Weight prep: fp32 -> bf16, transposing W1 ([H,F]->[F,H]) and W2 ([F,H]->[H,F])
// ---------------------------------------------------------------------------
__global__ __launch_bounds__(256) void prep_weights_kernel(
        const float* __restrict__ Wo, const float* __restrict__ W1,
        const float* __restrict__ W2, unsigned short* __restrict__ owb,
        unsigned short* __restrict__ w1t, unsigned short* __restrict__ w2t) {
    int i = blockIdx.x * 256 + threadIdx.x;
    const int n0 = 2 * H_ * H_;
    const int n1 = H_ * F_;
    if (i < n0) owb[i] = f2bf(Wo[i]);
    if (i < n1) {
        int f = i / H_, h = i - f * H_;
        w1t[i] = f2bf(W1[(size_t)h * F_ + f]);
        int h2 = i / F_, f2 = i - h2 * F_;
        w2t[i] = f2bf(W2[(size_t)f2 * H_ + h2]);
    }
}

// ---------------------------------------------------------------------------
// S4D discretization: modes[h*N+n] = (dr, di, 2*Re(Ct), 2*Im(Ct))
// ---------------------------------------------------------------------------
__global__ __launch_bounds__(256) void modes_kernel(
        const float* __restrict__ log_dt, const float* __restrict__ log_A_real,
        const float* __restrict__ A_imag, const float* __restrict__ C_real,
        const float* __restrict__ C_imag, float4* __restrict__ modes) {
    int i = blockIdx.x * 256 + threadIdx.x;
    if (i >= H_ * N_) return;
    int h = i >> 6;
    float dt = expf(log_dt[h]);
    float Ar = -expf(log_A_real[i]);
    float Ai = A_imag[i];
    float dr = dt * Ar, di = dt * Ai;
    float er = expf(dr);
    float wr = er * cosf(di), wi = er * sinf(di);
    float den = 1.0f / (Ar * Ar + Ai * Ai);
    float tr = ((wr - 1.0f) * Ar + wi * Ai) * den;
    float ti = (wi * Ar - (wr - 1.0f) * Ai) * den;
    float cr = C_real[i], ci = C_imag[i];
    modes[i] = make_float4(dr, di, 2.0f * (cr * tr - ci * ti),
                                   2.0f * (cr * ti + ci * tr));
}

// ---------------------------------------------------------------------------
// Per-h tables for the WMMA chunked S4D conv (B-fragment orientation, bf16).
// ---------------------------------------------------------------------------
__global__ __launch_bounds__(256) void s4d_tables_kernel(
        const float4* __restrict__ modes, unsigned short* __restrict__ tbt,
        unsigned short* __restrict__ wlt, unsigned short* __restrict__ gtt,
        float2* __restrict__ wt) {
    __shared__ float dr_s[N_], di_s[N_], z_s[N_], w_s[N_], K_s[T_];
    int h = blockIdx.x;
    int tid = threadIdx.x;
    if (tid < N_) {
        float4 m = modes[h * N_ + tid];
        dr_s[tid] = m.x; di_s[tid] = m.y; z_s[tid] = m.z; w_s[tid] = m.w;
    }
    __syncthreads();
    if (tid < T_) {
        float t = (float)tid;
        float acc = 0.f;
        for (int n = 0; n < N_; ++n) {
            float er = expf(t * dr_s[n]);
            float re = er * cosf(t * di_s[n]);
            float im = er * sinf(t * di_s[n]);
            acc += z_s[n] * re - w_s[n] * im;
        }
        K_s[tid] = acc;
    }
    if (tid < N_) {
        float er = expf(64.f * dr_s[tid]);
        wt[h * N_ + tid] = make_float2(er * cosf(64.f * di_s[tid]),
                                       er * sinf(64.f * di_s[tid]));
    }
    __syncthreads();
    for (int idx = tid; idx < T_ * T_; idx += 256) {
        int i = idx >> 6, j = idx & 63;
        tbt[((size_t)h * T_ + i) * T_ + j] = f2bf((i >= j) ? K_s[i - j] : 0.f);
    }
    for (int idx = tid; idx < 2 * N_ * T_; idx += 256) {
        int nc = idx >> 6, j = idx & 63;
        int n = nc >> 1;
        float p = (float)(63 - j);
        float er = expf(p * dr_s[n]);
        float v = (nc & 1) ? er * sinf(p * di_s[n]) : er * cosf(p * di_s[n]);
        wlt[((size_t)h * 2 * N_ + nc) * T_ + j] = f2bf(v);
    }
    for (int idx = tid; idx < T_ * 2 * N_; idx += 256) {
        int i = idx >> 7, k = idx & 127;
        int n = k >> 1;
        float p = (float)(i + 1);
        float er = expf(p * dr_s[n]);
        float re = er * cosf(p * di_s[n]);
        float im = er * sinf(p * di_s[n]);
        float v = (k & 1) ? -(z_s[n] * im + w_s[n] * re)
                          :  (z_s[n] * re - w_s[n] * im);
        gtt[((size_t)h * T_ + i) * 2 * N_ + k] = f2bf(v);
    }
}

// ---------------------------------------------------------------------------
// LN1 + layout transpose: x[B,L,H] -> xt[B,H,L] and ut[B,H,L] (LN'd)
// ---------------------------------------------------------------------------
__global__ __launch_bounds__(256) void ln1_kernel(
        const float* __restrict__ x, const float* __restrict__ g,
        const float* __restrict__ bb, float* __restrict__ xt,
        float* __restrict__ ut) {
    __shared__ float tile[16][H_ + 1];
    __shared__ float red[16][16], red2[16][16];
    __shared__ float mu_s[16], rs_s[16];
    int tid = threadIdx.x;
    int row = tid >> 4;
    int cg  = tid & 15;
    int bidx = blockIdx.x;
    int b  = bidx >> 7;
    int l0 = (bidx & 127) * 16;
    const float* xp = x + ((size_t)b * L_ + (l0 + row)) * H_;
    float s = 0.f, ss = 0.f;
    #pragma unroll
    for (int j = 0; j < 32; j += 4) {
        float4 v = *(const float4*)(xp + cg * 32 + j);
        int c = cg * 32 + j;
        tile[row][c + 0] = v.x; tile[row][c + 1] = v.y;
        tile[row][c + 2] = v.z; tile[row][c + 3] = v.w;
        s += v.x + v.y + v.z + v.w;
        ss += v.x * v.x + v.y * v.y + v.z * v.z + v.w * v.w;
    }
    red[row][cg] = s; red2[row][cg] = ss;
    __syncthreads();
    if (cg == 0) {
        float S = 0.f, SS = 0.f;
        #pragma unroll
        for (int j = 0; j < 16; ++j) { S += red[row][j]; SS += red2[row][j]; }
        float mu = S * (1.0f / H_);
        mu_s[row] = mu;
        rs_s[row] = rsqrtf(SS * (1.0f / H_) - mu * mu + 1e-5f);
    }
    __syncthreads();
    int lw = tid & 15;
    int hg = tid >> 4;
    for (int it = 0; it < 32; ++it) {
        int h = hg + it * 16;
        float v = tile[lw][h];
        size_t idx = ((size_t)b * H_ + h) * L_ + l0 + lw;
        xt[idx] = v;
        ut[idx] = (v - mu_s[lw]) * rs_s[lw] * g[h] + bb[h];
    }
}

// ---------------------------------------------------------------------------
// WMMA-based chunked S4D conv + D-skip + GELU (one 2-wave block per (b,h)).
// ---------------------------------------------------------------------------
__global__ __launch_bounds__(64) void s4d_conv_kernel(
        const float* __restrict__ ut, const unsigned short* __restrict__ tbt,
        const unsigned short* __restrict__ wlt, const unsigned short* __restrict__ gtt,
        const float2* __restrict__ wt, const float* __restrict__ D,
        unsigned short* __restrict__ yt) {
    __shared__ unsigned short U_lds[C_][72];
    __shared__ float          E_lds[C_][132];
    __shared__ unsigned short S_lds[C_][136];
    int tid = threadIdx.x, lane = tid & 31, wv = tid >> 5;
    int bh = blockIdx.x;
    int h = bh & (H_ - 1);
    const float* up = ut + (size_t)bh * L_;
    unsigned short* yp = yt + (size_t)bh * L_;
    float Dh = D[h];

    #pragma unroll
    for (int it = 0; it < 8; ++it) {
        int l = it * 256 + tid * 4;
        float4 v = *(const float4*)(up + l);
        unsigned short p[4] = { f2bf(v.x), f2bf(v.y), f2bf(v.z), f2bf(v.w) };
        *(uint2*)&U_lds[l >> 6][l & 63] = *(uint2*)p;
    }
    __syncthreads();

    int mrow = wv * 16 + (lane & 15);
    int kb = (lane < 16) ? 0 : 16;
    int kh = (lane < 16) ? 0 : 16;
    v8f z8 = {0.f, 0.f, 0.f, 0.f, 0.f, 0.f, 0.f, 0.f};

    {   // step 2: chunk-end states E (N=128, K=64)
        v8f acc[8] = {z8, z8, z8, z8, z8, z8, z8, z8};
        const unsigned short* Wl = wlt + (size_t)h * 2 * N_ * T_;
        #pragma unroll
        for (int ks = 0; ks < 2; ++ks) {
            int k0 = ks * 32;
            Frag16 a;
            a.u[0] = *(const uint4*)&U_lds[mrow][k0 + kb];
            a.u[1] = *(const uint4*)&U_lds[mrow][k0 + kb + 8];
            #pragma unroll
            for (int j = 0; j < 8; ++j) {
                Frag16 bf;
                const unsigned short* r = Wl + (size_t)(j * 16 + (lane & 15)) * T_ + k0 + kh;
                bf.u[0] = *(const uint4*)r;
                bf.u[1] = *(const uint4*)(r + 8);
                acc[j] = __builtin_amdgcn_wmma_f32_16x16x32_bf16(
                             false, a.v, false, bf.v, (short)0, acc[j], false, false);
            }
        }
        int mofs = (lane >> 4) * 8;
        #pragma unroll
        for (int j = 0; j < 8; ++j) {
            int nc = j * 16 + (lane & 15);
            #pragma unroll
            for (int i = 0; i < 8; ++i)
                E_lds[wv * 16 + mofs + i][nc] = acc[j][i];
        }
    }
    __syncthreads();

    {   // step 3: serial prefix per complex mode
        float2 wT = wt[h * N_ + tid];
        float sre = 0.f, sim = 0.f;
        #pragma unroll 4
        for (int c = 0; c < C_; ++c) {
            S_lds[c][2 * tid]     = f2bf(sre);
            S_lds[c][2 * tid + 1] = f2bf(sim);
            float er = E_lds[c][2 * tid], ei = E_lds[c][2 * tid + 1];
            float nr = wT.x * sre - wT.y * sim + er;
            float ni = wT.x * sim + wT.y * sre + ei;
            sre = nr; sim = ni;
        }
    }
    __syncthreads();

    {   // steps 1 + 4: Y = U x Toep^T + S x G^T
        v8f acc[4] = {z8, z8, z8, z8};
        const unsigned short* Tb = tbt + (size_t)h * T_ * T_;
        #pragma unroll
        for (int ks = 0; ks < 2; ++ks) {
            int k0 = ks * 32;
            Frag16 a;
            a.u[0] = *(const uint4*)&U_lds[mrow][k0 + kb];
            a.u[1] = *(const uint4*)&U_lds[mrow][k0 + kb + 8];
            #pragma unroll
            for (int j = 0; j < 4; ++j) {
                Frag16 bf;
                const unsigned short* r = Tb + (size_t)(j * 16 + (lane & 15)) * T_ + k0 + kh;
                bf.u[0] = *(const uint4*)r;
                bf.u[1] = *(const uint4*)(r + 8);
                acc[j] = __builtin_amdgcn_wmma_f32_16x16x32_bf16(
                             false, a.v, false, bf.v, (short)0, acc[j], false, false);
            }
        }
        const unsigned short* Gt = gtt + (size_t)h * T_ * 2 * N_;
        #pragma unroll
        for (int ks = 0; ks < 4; ++ks) {
            int k0 = ks * 32;
            Frag16 a;
            a.u[0] = *(const uint4*)&S_lds[mrow][k0 + kb];
            a.u[1] = *(const uint4*)&S_lds[mrow][k0 + kb + 8];
            #pragma unroll
            for (int j = 0; j < 4; ++j) {
                Frag16 bf;
                const unsigned short* r = Gt + (size_t)(j * 16 + (lane & 15)) * 2 * N_ + k0 + kh;
                bf.u[0] = *(const uint4*)r;
                bf.u[1] = *(const uint4*)(r + 8);
                acc[j] = __builtin_amdgcn_wmma_f32_16x16x32_bf16(
                             false, a.v, false, bf.v, (short)0, acc[j], false, false);
            }
        }
        int mofs = (lane >> 4) * 8;
        #pragma unroll
        for (int j = 0; j < 4; ++j) {
            int n = j * 16 + (lane & 15);
            #pragma unroll
            for (int i = 0; i < 8; ++i) {
                int c = wv * 16 + mofs + i;
                int l = c * T_ + n;
                float y = acc[j][i] + Dh * up[l];
                yp[l] = f2bf(gelu_exact(y));
            }
        }
    }
}

// ---------------------------------------------------------------------------
// GEMM helpers (bf16 WMMA 16x16x32, fp32 accum). Block = 256 thr = 8 waves.
// A tiles staged by async global->LDS DMA; B tiles transposed through VGPRs.
// Double-buffered: one barrier per K-step.
// ---------------------------------------------------------------------------
__device__ __forceinline__ void stage_b_tile(const unsigned short* __restrict__ Bsrc,
        int k0, int n0, unsigned short (*BT)[40], int tid) {
    int k = tid >> 3;
    int n = (tid & 7) * 8;
    uint4 v = *(const uint4*)(Bsrc + (size_t)(k0 + k) * L_ + n0 + n);
    unsigned short tmp[8];
    *(uint4*)tmp = v;
    #pragma unroll
    for (int j = 0; j < 8; ++j) BT[n + j][k] = tmp[j];
}

__device__ __forceinline__ void wave_mma(const unsigned short (*Al)[40], int mrow0,
        const unsigned short (*BT)[40], int lane, v8f acc[4]) {
    Frag16 a;
    int m = mrow0 + (lane & 15);
    int kb = (lane < 16) ? 0 : 16;
    a.u[0] = *(const uint4*)&Al[m][kb];
    a.u[1] = *(const uint4*)&Al[m][kb + 8];
    int kh = (lane < 16) ? 0 : 16;
    #pragma unroll
    for (int j = 0; j < 4; ++j) {
        Frag16 bf;
        int n = j * 16 + (lane & 15);
        bf.u[0] = *(const uint4*)&BT[n][kh];
        bf.u[1] = *(const uint4*)&BT[n][kh + 8];
        acc[j] = __builtin_amdgcn_wmma_f32_16x16x32_bf16(
                     false, a.v, false, bf.v, (short)0, acc[j], false, false);
    }
}

// ---------------------------------------------------------------------------
// GEMM1 + GLU + residual: z = out_W[1024,512] x yt_b[512,L]; x2t = xt + GLU(z)
// ---------------------------------------------------------------------------
__global__ __launch_bounds__(256) void gemm_glu_kernel(
        const unsigned short* __restrict__ Wo, const unsigned short* __restrict__ ytb,
        const float* __restrict__ out_b, const float* __restrict__ xt,
        float* __restrict__ x2t) {
    __shared__ unsigned short Al[2][256][40];
    __shared__ unsigned short BT[2][64][40];
    int b = blockIdx.z;
    int m0 = blockIdx.y * 128;
    int n0 = blockIdx.x * 64;
    const unsigned short* Bsrc = ytb + (size_t)b * H_ * L_;
    int tid = threadIdx.x, lane = tid & 31, wv = tid >> 5;
    v8f z8 = {0.f, 0.f, 0.f, 0.f, 0.f, 0.f, 0.f, 0.f};
    v8f accA[4] = {z8, z8, z8, z8};
    v8f accG[4] = {z8, z8, z8, z8};

    auto issue_a = [&](int buf, int k0) {
        #pragma unroll
        for (int it = 0; it < 4; ++it) {
            int r = (tid >> 2) + it * 64;
            int kk = (tid & 3) * 8;
            int gr = (r < 128) ? (m0 + r) : (H_ + m0 + (r - 128));
            async_copy_b128(&Al[buf][r][kk], Wo + (size_t)gr * H_ + k0 + kk);
        }
    };
    issue_a(0, 0);
    stage_b_tile(Bsrc, 0, n0, BT[0], tid);
    async_wait0();
    __syncthreads();
    for (int ks = 0; ks < H_ / 32; ++ks) {
        int cur = ks & 1;
        int k0n = (ks + 1) * 32;
        if (k0n < H_) {
            issue_a(cur ^ 1, k0n);
            stage_b_tile(Bsrc, k0n, n0, BT[cur ^ 1], tid);
        }
        wave_mma(Al[cur], wv * 16, BT[cur], lane, accA);
        wave_mma(Al[cur], 128 + wv * 16, BT[cur], lane, accG);
        async_wait0();
        __syncthreads();
    }
    int mofs = (lane >> 4) * 8;
    #pragma unroll
    for (int j = 0; j < 4; ++j) {
        int n = n0 + j * 16 + (lane & 15);
        #pragma unroll
        for (int i = 0; i < 8; ++i) {
            int m = m0 + wv * 16 + mofs + i;
            float a = accA[j][i] + out_b[m];
            float gg = accG[j][i] + out_b[H_ + m];
            float y = a * (1.0f / (1.0f + __expf(-gg)));
            size_t idx = ((size_t)b * H_ + m) * L_ + n;
            x2t[idx] = xt[idx] + y;
        }
    }
}

// ---------------------------------------------------------------------------
// LN2 over the strided H dim of x2t[B,H,L]
// ---------------------------------------------------------------------------
__global__ __launch_bounds__(256) void ln2_stats_kernel(
        const float* __restrict__ x2t, float* __restrict__ mu, float* __restrict__ rs) {
    int b = blockIdx.y;
    int l = blockIdx.x * 256 + threadIdx.x;
    const float* p = x2t + (size_t)b * H_ * L_ + l;
    float s = 0.f, ss = 0.f;
    for (int h = 0; h < H_; ++h) { float v = p[(size_t)h * L_]; s += v; ss += v * v; }
    float m = s * (1.0f / H_);
    mu[b * L_ + l] = m;
    rs[b * L_ + l] = rsqrtf(ss * (1.0f / H_) - m * m + 1e-5f);
}

__global__ __launch_bounds__(256) void ln2_norm_kernel(
        const float* __restrict__ x2t, const float* __restrict__ mu,
        const float* __restrict__ rs, const float* __restrict__ g,
        const float* __restrict__ bb, unsigned short* __restrict__ h2t) {
    size_t i = (size_t)blockIdx.x * 256 + threadIdx.x;
    int l = (int)(i & (L_ - 1));
    int h = (int)((i >> 11) & (H_ - 1));
    int b = (int)(i >> 20);
    float v = (x2t[i] - mu[b * L_ + l]) * rs[b * L_ + l] * g[h] + bb[h];
    h2t[i] = f2bf(v);
}

// ---------------------------------------------------------------------------
// Generic GEMM: C = A[M,K] x B_b[K,L]. EPI 0: +b1, GELU -> bf16.
// EPI 1: +b2 + residual -> fp32 out[B,L,H] (transposed final write).
// ---------------------------------------------------------------------------
template<int EPI>
__global__ __launch_bounds__(256) void gemm_kernel(
        const unsigned short* __restrict__ A, int M, int K,
        const unsigned short* __restrict__ Ball, const float* __restrict__ bias,
        const float* __restrict__ res, float* __restrict__ outf,
        unsigned short* __restrict__ outb) {
    __shared__ unsigned short Al[2][128][40];
    __shared__ unsigned short BT[2][64][40];
    int b = blockIdx.z;
    int m0 = blockIdx.y * 128;
    int n0 = blockIdx.x * 64;
    const unsigned short* Bsrc = Ball + (size_t)b * K * L_;
    int tid = threadIdx.x, lane = tid & 31, wv = tid >> 5;
    v8f z8 = {0.f, 0.f, 0.f, 0.f, 0.f, 0.f, 0.f, 0.f};
    v8f acc[4] = {z8, z8, z8, z8};

    auto issue_a = [&](int buf, int k0) {
        #pragma unroll
        for (int it = 0; it < 2; ++it) {
            int r = (tid >> 2) + it * 64;
            int kk = (tid & 3) * 8;
            async_copy_b128(&Al[buf][r][kk], A + (size_t)(m0 + r) * K + k0 + kk);
        }
    };
    issue_a(0, 0);
    stage_b_tile(Bsrc, 0, n0, BT[0], tid);
    async_wait0();
    __syncthreads();
    int nks = K / 32;
    for (int ks = 0; ks < nks; ++ks) {
        int cur = ks & 1;
        int k0n = (ks + 1) * 32;
        if (k0n < K) {
            issue_a(cur ^ 1, k0n);
            stage_b_tile(Bsrc, k0n, n0, BT[cur ^ 1], tid);
        }
        wave_mma(Al[cur], wv * 16, BT[cur], lane, acc);
        async_wait0();
        __syncthreads();
    }
    int mofs = (lane >> 4) * 8;
    #pragma unroll
    for (int j = 0; j < 4; ++j) {
        int n = n0 + j * 16 + (lane & 15);
        #pragma unroll
        for (int i = 0; i < 8; ++i) {
            int m = m0 + wv * 16 + mofs + i;
            float v = acc[j][i] + bias[m];
            if (EPI == 0) {
                outb[((size_t)b * M + m) * L_ + n] = f2bf(gelu_exact(v));
            } else {
                v += res[((size_t)b * H_ + m) * L_ + n];
                outf[((size_t)b * L_ + n) * H_ + m] = v;
            }
        }
    }
}

// ---------------------------------------------------------------------------
// Launch
// ---------------------------------------------------------------------------
extern "C" void kernel_launch(void* const* d_in, const int* in_sizes, int n_in,
                              void* d_out, int out_size, void* d_ws, size_t ws_size,
                              hipStream_t stream) {
    const float* x          = (const float*)d_in[0];
    const float* ln1_g      = (const float*)d_in[1];
    const float* ln1_b      = (const float*)d_in[2];
    const float* log_dt     = (const float*)d_in[3];
    const float* log_A_real = (const float*)d_in[4];
    const float* A_imag     = (const float*)d_in[5];
    const float* C_real     = (const float*)d_in[6];
    const float* C_imag     = (const float*)d_in[7];
    const float* Dp         = (const float*)d_in[8];
    const float* out_W      = (const float*)d_in[9];
    const float* out_b      = (const float*)d_in[10];
    const float* ln2_g      = (const float*)d_in[11];
    const float* ln2_b      = (const float*)d_in[12];
    const float* W1         = (const float*)d_in[13];
    const float* b1         = (const float*)d_in[14];
    const float* W2         = (const float*)d_in[15];
    const float* b2         = (const float*)d_in[16];
    (void)in_sizes; (void)n_in; (void)out_size; (void)ws_size;

    char* ws = (char*)d_ws;
    size_t off = 0;
    auto take = [&](size_t bytes) {
        char* p = ws + off;
        off += (bytes + 255) & ~(size_t)255;
        return p;
    };
    const size_t BHL = (size_t)B_ * H_ * L_;
    float4*         modes = (float4*)take((size_t)H_ * N_ * 16);
    unsigned short* owb   = (unsigned short*)take((size_t)2 * H_ * H_ * 2);
    unsigned short* w1t   = (unsigned short*)take((size_t)H_ * F_ * 2);
    unsigned short* w2t   = (unsigned short*)take((size_t)F_ * H_ * 2);
    unsigned short* tbt   = (unsigned short*)take((size_t)H_ * T_ * T_ * 2);
    unsigned short* wlt   = (unsigned short*)take((size_t)H_ * 2 * N_ * T_ * 2);
    unsigned short* gtt   = (unsigned short*)take((size_t)H_ * T_ * 2 * N_ * 2);
    float2*         wt    = (float2*)take((size_t)H_ * N_ * 8);
    float*          xt    = (float*)take(BHL * 4);
    float*          ut    = (float*)take(BHL * 4);
    unsigned short* yt    = (unsigned short*)take(BHL * 2);
    float*          x2t   = (float*)take(BHL * 4);
    float*          mu    = (float*)take((size_t)B_ * L_ * 4);
    float*          rs    = (float*)take((size_t)B_ * L_ * 4);
    unsigned short* h2t   = (unsigned short*)take(BHL * 2);
    unsigned short* mt    = (unsigned short*)take((size_t)B_ * F_ * L_ * 2);

    prep_weights_kernel<<<(H_ * F_) / 256, 256, 0, stream>>>(out_W, W1, W2, owb, w1t, w2t);
    modes_kernel<<<(H_ * N_) / 256, 256, 0, stream>>>(log_dt, log_A_real, A_imag,
                                                      C_real, C_imag, modes);
    s4d_tables_kernel<<<H_, 256, 0, stream>>>(modes, tbt, wlt, gtt, wt);
    ln1_kernel<<<(B_ * L_) / 16, 256, 0, stream>>>(x, ln1_g, ln1_b, xt, ut);
    s4d_conv_kernel<<<B_ * H_, 64, 0, stream>>>(ut, tbt, wlt, gtt, wt, Dp, yt);
    gemm_glu_kernel<<<dim3(L_ / 64, H_ / 128, B_), 256, 0, stream>>>(owb, yt, out_b, xt, x2t);
    ln2_stats_kernel<<<dim3(L_ / 256, B_), 256, 0, stream>>>(x2t, mu, rs);
    ln2_norm_kernel<<<(int)(BHL / 256), 256, 0, stream>>>(x2t, mu, rs, ln2_g, ln2_b, h2t);
    gemm_kernel<0><<<dim3(L_ / 64, F_ / 128, B_), 256, 0, stream>>>(
        w1t, F_, H_, h2t, b1, nullptr, nullptr, mt);
    gemm_kernel<1><<<dim3(L_ / 64, H_ / 128, B_), 256, 0, stream>>>(
        w2t, H_, F_, mt, b2, x2t, (float*)d_out, nullptr);
}
